// SOCA_41394894799117
// MI455X (gfx1250) — matrix operand order repaired and checked by reading, not compiled, not run on hardware
//
#include <hip/hip_runtime.h>
#include <hip/hip_bf16.h>
#include <math.h>

#define BATCH 32
#define CCH   256
#define MPIX  9216          // 96*96
#define MATN  65536         // 256*256

typedef __attribute__((ext_vector_type(16))) __bf16 v16bf;
typedef __attribute__((ext_vector_type(8)))  float  v8f;
typedef __attribute__((ext_vector_type(2)))  float  v2f;

__device__ __forceinline__ __bf16 f2bf(float f) {
  union { float f; unsigned u; } v; v.f = f;
  unsigned r = (v.u + 0x7FFFu + ((v.u >> 16) & 1u)) >> 16;
  union { unsigned short s; __bf16 b; } o; o.s = (unsigned short)r;
  return o.b;
}

// ---------------------------------------------------------------------------
// 1) Per-(b,c) mean over M pixels. One wave (32 lanes) per row.
// ---------------------------------------------------------------------------
__global__ void mean_kernel(const float* __restrict__ x, float* __restrict__ means) {
  int wave = threadIdx.x >> 5, lane = threadIdx.x & 31;
  int row  = blockIdx.x * 8 + wave;            // row in [0, B*C)
  const float4* xr = (const float4*)(x + (size_t)row * MPIX);
  float s = 0.f;
  for (int i = lane; i < MPIX / 4; i += 32) {
    float4 v = xr[i];
    s += v.x + v.y + v.z + v.w;
  }
  #pragma unroll
  for (int o = 16; o > 0; o >>= 1) s += __shfl_xor(s, o, 32);
  if (lane == 0) means[row] = s * (1.0f / MPIX);
}

// ---------------------------------------------------------------------------
// 2) covpool: cov[b] = (1/M) * xc[b] @ xc[b]^T via bf16 WMMA, fp32 accum.
//    Grid (2,2,B): 128x128 cov tile per WG.  Block = 256 thr = 8 waves,
//    each wave computes a 32x64 sub-tile (2x4 of 16x16 accumulators).
// ---------------------------------------------------------------------------
__global__ void covpool_kernel(const float* __restrict__ x,
                               const float* __restrict__ means,
                               float* __restrict__ cov) {
  const int LDA = 34;  // bf16 stride: 17 banks, coprime with 64 -> conflict free
  __shared__ __bf16 Asm[128 * LDA];
  __shared__ __bf16 Bsm[128 * LDA];
  __shared__ float  mA[128], mB[128];

  int b = blockIdx.z;
  int rowBase = blockIdx.y * 128;
  int colBase = blockIdx.x * 128;
  int tid  = threadIdx.x;
  int wave = tid >> 5, lane = tid & 31;
  int wr = wave >> 1;   // rows wr*32
  int wc = wave & 1;    // cols wc*64

  if (tid < 128) mA[tid]       = means[b * CCH + rowBase + tid];
  else           mB[tid - 128] = means[b * CCH + colBase + (tid - 128)];

  v8f acc[2][4];
  const v8f vzero = {0.f,0.f,0.f,0.f,0.f,0.f,0.f,0.f};
  #pragma unroll
  for (int ri = 0; ri < 2; ++ri)
    #pragma unroll
    for (int ci = 0; ci < 4; ++ci) acc[ri][ci] = vzero;

  const float* xb = x + (size_t)b * CCH * MPIX;
  int Mi = lane & 15, half = lane >> 4;

  for (int k0 = 0; k0 < MPIX; k0 += 32) {
    __syncthreads();   // previous fragment reads done (also covers mA/mB init)
    // Stage centered bf16 tiles: 128 channels x 32 pixels each.
    #pragma unroll
    for (int i = 0; i < 4; ++i) {
      int v   = tid + i * 256;        // 1024 float4 loads
      int ch  = v >> 3;               // 8 float4 per channel
      int off = (v & 7) * 4;
      float4 d = *(const float4*)(xb + (size_t)(rowBase + ch) * MPIX + k0 + off);
      float m = mA[ch];
      __bf16* p = &Asm[ch * LDA + off];
      p[0] = f2bf(d.x - m); p[1] = f2bf(d.y - m);
      p[2] = f2bf(d.z - m); p[3] = f2bf(d.w - m);
    }
    #pragma unroll
    for (int i = 0; i < 4; ++i) {
      int v   = tid + i * 256;
      int ch  = v >> 3;
      int off = (v & 7) * 4;
      float4 d = *(const float4*)(xb + (size_t)(colBase + ch) * MPIX + k0 + off);
      float m = mB[ch];
      __bf16* p = &Bsm[ch * LDA + off];
      p[0] = f2bf(d.x - m); p[1] = f2bf(d.y - m);
      p[2] = f2bf(d.z - m); p[3] = f2bf(d.w - m);
    }
    __syncthreads();

    // A fragment, 16x32 bf16 (ISA 7.12.2): lane M=lane%16, half=lane/16
    //   VGPR j<4 : K = half*8 + 2j {+0,+1};  VGPR j>=4: K = 16 + half*8 + 2(j-4)
    v16bf af[2], bfr[4];
    #pragma unroll
    for (int ri = 0; ri < 2; ++ri) {
      int row = wr * 32 + ri * 16 + Mi;
      #pragma unroll
      for (int e = 0; e < 16; ++e) {
        int j = e >> 1, sel = e & 1;
        int K = (j < 4 ? half * 8 + j * 2 : 16 + half * 8 + (j - 4) * 2) + sel;
        af[ri][e] = Asm[row * LDA + K];
      }
    }
    // B fragment, 32x16 bf16: B[k][n] = xc[colChan=n][k].
//   VGPR j<4 : K = half*8 + 2j {+0,+1};  VGPR j>=4: K = 16 + half*8 + 2(j-4)
    #pragma unroll
    for (int ci = 0; ci < 4; ++ci) {
      int col = wc * 64 + ci * 16 + Mi;   // N index
      #pragma unroll
      for (int e = 0; e < 16; ++e) {
        int j = e >> 1, sel = e & 1;
        int K = (j < 4 ? half * 8 + j * 2 : 16 + half * 8 + (j - 4) * 2) + sel;
        bfr[ci][e] = Bsm[col * LDA + K];
      }
    }
    #pragma unroll
    for (int ri = 0; ri < 2; ++ri)
      #pragma unroll
      for (int ci = 0; ci < 4; ++ci)
        acc[ri][ci] = __builtin_amdgcn_wmma_f32_16x16x32_bf16(
            false, af[ri], false, bfr[ci], (short)0, acc[ri][ci], false, false);
  }

  // C/D layout: lane N=lane%16, rowgrp=lane/16; VGPR r -> M = r + 8*rowgrp
  int N = lane & 15, rg = lane >> 4;
  const float invM = 1.0f / MPIX;
  float* covb = cov + (size_t)b * MATN;
  #pragma unroll
  for (int ri = 0; ri < 2; ++ri)
    #pragma unroll
    for (int ci = 0; ci < 4; ++ci)
      #pragma unroll
      for (int r = 0; r < 8; ++r) {
        int grow = rowBase + wr * 32 + ri * 16 + r + 8 * rg;
        int gcol = colBase + wc * 64 + ci * 16 + N;
        covb[grow * CCH + gcol] = acc[ri][ci][r] * invM;
      }
}

// ---------------------------------------------------------------------------
// 3) trace + prep for Newton-Schulz
// ---------------------------------------------------------------------------
__global__ void trace_kernel(const float* __restrict__ cov,
                             float* __restrict__ normA, float* __restrict__ sqrtn) {
  __shared__ float red[256];
  int b = blockIdx.x, tid = threadIdx.x;
  red[tid] = cov[(size_t)b * MATN + tid * 257];   // diagonal
  __syncthreads();
  for (int s = 128; s > 0; s >>= 1) {
    if (tid < s) red[tid] += red[tid + s];
    __syncthreads();
  }
  if (tid == 0) {
    float t = red[0];
    normA[b] = t;
    sqrtn[b] = sqrtf(fmaxf(t, 1e-20f));
  }
}

__global__ void prep_kernel(const float* __restrict__ cov, const float* __restrict__ normA,
                            float* __restrict__ Ahat, float* __restrict__ Z0) {
  int idx = blockIdx.x * 256 + threadIdx.x;   // [0, B*MATN)
  int b = idx >> 16;
  int rc = idx & 65535;
  int r = rc >> 8, c = rc & 255;
  float a = cov[idx] / normA[b];
  Ahat[idx] = a;
  Z0[idx]   = (r == c ? 1.5f : 0.0f) - 0.5f * a;
}

// ---------------------------------------------------------------------------
// 4) Batched fp32 256x256x256 GEMM using v_wmma_f32_16x16x4_f32.
//    Grid (4,4,B): 64x64 tile / WG; block = 128 thr = 4 waves,
//    each wave a 16x64 strip (4 accumulators).  Fused NS epilogues.
// ---------------------------------------------------------------------------
#define NS_PLAIN 0
#define NS_F     1   // D = 1.5I - 0.5*(A@B)
#define NS_SCALE 2   // D = (A@B) * scale[b]

__global__ void ns_gemm_kernel(const float* __restrict__ A, const float* __restrict__ Bm,
                               float* __restrict__ D, int mode,
                               const float* __restrict__ scale) {
  const int LDA = 17;  // odd fp32 stride -> conflict free
  __shared__ float As[64 * LDA];
  __shared__ float Bs[16 * 64];

  int b = blockIdx.z;
  int rowBase = blockIdx.y * 64;
  int colBase = blockIdx.x * 64;
  int tid = threadIdx.x;             // 128
  int wave = tid >> 5, lane = tid & 31;
  const float* Ab = A  + (size_t)b * MATN;
  const float* Bb = Bm + (size_t)b * MATN;

  v8f acc[4];
  const v8f vzero = {0.f,0.f,0.f,0.f,0.f,0.f,0.f,0.f};
  #pragma unroll
  for (int ci = 0; ci < 4; ++ci) acc[ci] = vzero;

  int Mi = lane & 15, half = lane >> 4;

  for (int k0 = 0; k0 < 256; k0 += 16) {
    __syncthreads();
    // Stage A 64x16 (row-major, padded) and B 16x64 (row-major).
    #pragma unroll
    for (int i = 0; i < 2; ++i) {
      int v = tid + i * 128;          // 256 float4
      int r = v >> 2, off = (v & 3) * 4;
      float4 d = *(const float4*)(Ab + (rowBase + r) * 256 + k0 + off);
      float* p = &As[r * LDA + off];
      p[0] = d.x; p[1] = d.y; p[2] = d.z; p[3] = d.w;
    }
    #pragma unroll
    for (int i = 0; i < 2; ++i) {
      int v = tid + i * 128;
      int k = v >> 4, off = (v & 15) * 4;
      float4 d = *(const float4*)(Bb + (k0 + k) * 256 + colBase + off);
      float* p = &Bs[k * 64 + off];
      p[0] = d.x; p[1] = d.y; p[2] = d.z; p[3] = d.w;
    }
    __syncthreads();

    #pragma unroll
    for (int kk = 0; kk < 16; kk += 4) {
      // A frag 16x4 f32: VGPR0 K=2*half, VGPR1 K=2*half+1 (ISA 7.12.2)
      v2f a;
      a[0] = As[(wave * 16 + Mi) * LDA + kk + 2 * half + 0];
      a[1] = As[(wave * 16 + Mi) * LDA + kk + 2 * half + 1];
      #pragma unroll
      for (int ci = 0; ci < 4; ++ci) {
        // B frag 4x16 f32, mirrored lane layout
        v2f bb;
        bb[0] = Bs[(kk + 2 * half + 0) * 64 + ci * 16 + Mi];
        bb[1] = Bs[(kk + 2 * half + 1) * 64 + ci * 16 + Mi];
        acc[ci] = __builtin_amdgcn_wmma_f32_16x16x4_f32(
            false, a, false, bb, (short)0, acc[ci], false, false);
      }
    }
  }

  int N = lane & 15, rg = lane >> 4;
  float sc = (mode == NS_SCALE) ? scale[b] : 1.0f;
  float* Db = D + (size_t)b * MATN;
  #pragma unroll
  for (int ci = 0; ci < 4; ++ci)
    #pragma unroll
    for (int r = 0; r < 8; ++r) {
      int grow = rowBase + wave * 16 + r + 8 * rg;
      int gcol = colBase + ci * 16 + N;
      float p = acc[ci][r];
      float o = (mode == NS_F) ? ((grow == gcol ? 1.5f : 0.0f) - 0.5f * p)
                               : p * sc;
      Db[grow * 256 + gcol] = o;
    }
}

// ---------------------------------------------------------------------------
// 5) column-mean of cov_sqrt + squeeze-excite MLP -> per-(b,c) gate
// ---------------------------------------------------------------------------
__global__ void semlp_kernel(const float* __restrict__ S,
                             const float* __restrict__ w1, const float* __restrict__ b1,
                             const float* __restrict__ w2, const float* __restrict__ b2,
                             float* __restrict__ sgate) {
  __shared__ float v[256];
  __shared__ float h[32];
  int b = blockIdx.x, tid = threadIdx.x;
  const float* Sb = S + (size_t)b * MATN;
  float s = 0.f;
  for (int c = 0; c < 256; ++c) s += Sb[c * 256 + tid];   // mean over axis=1
  v[tid] = s * (1.0f / 256.0f);
  __syncthreads();
  if (tid < 32) {
    float acc = b1[tid];
    const float* wr = w1 + tid * 256;
    for (int c = 0; c < 256; ++c) acc += v[c] * wr[c];
    h[tid] = fmaxf(acc, 0.f);
  }
  __syncthreads();
  float acc = b2[tid];
  const float* wr = w2 + tid * 32;
  #pragma unroll
  for (int j = 0; j < 32; ++j) acc += h[j] * wr[j];
  sgate[b * 256 + tid] = 1.0f / (1.0f + expf(-acc));
}

// ---------------------------------------------------------------------------
// 6) out = gate[b,c] * x  (float4)
// ---------------------------------------------------------------------------
__global__ void scale_kernel(const float* __restrict__ x,
                             const float* __restrict__ sgate,
                             float* __restrict__ out) {
  int v = blockIdx.x * 256 + threadIdx.x;   // float4 index
  int bc = v / (MPIX / 4);                  // (b*C + c)
  float s = sgate[bc];
  float4 d = ((const float4*)x)[v];
  d.x *= s; d.y *= s; d.z *= s; d.w *= s;
  ((float4*)out)[v] = d;
}

// ---------------------------------------------------------------------------
extern "C" void kernel_launch(void* const* d_in, const int* in_sizes, int n_in,
                              void* d_out, int out_size, void* d_ws, size_t ws_size,
                              hipStream_t stream) {
  const float* x  = (const float*)d_in[0];
  const float* w1 = (const float*)d_in[1];
  const float* b1 = (const float*)d_in[2];
  const float* w2 = (const float*)d_in[3];
  const float* b2 = (const float*)d_in[4];
  float* out = (float*)d_out;

  const size_t MAT = (size_t)BATCH * MATN;     // 2,097,152 floats (8 MB)
  float* ws = (float*)d_ws;
  float* B0 = ws;            // cov
  float* B1 = B0 + MAT;      // Ahat
  float* B2 = B1 + MAT;      // Z0
  float* B3 = B2 + MAT;      // Y
  float* B4 = B3 + MAT;      // T (ZY)
  float* means = B4 + MAT;                     // B*C
  float* normA = means + (size_t)BATCH * CCH;  // B
  float* sqrtn = normA + BATCH;                // B
  float* sgate = sqrtn + BATCH;                // B*C

  mean_kernel   <<<BATCH * CCH / 8, 256, 0, stream>>>(x, means);
  covpool_kernel<<<dim3(2, 2, BATCH), 256, 0, stream>>>(x, means, B0);
  trace_kernel  <<<BATCH, 256, 0, stream>>>(B0, normA, sqrtn);
  prep_kernel   <<<(int)(MAT / 256), 256, 0, stream>>>(B0, normA, B1, B2);

  dim3 gg(4, 4, BATCH);
  // Y = Ahat @ Z0
  ns_gemm_kernel<<<gg, 128, 0, stream>>>(B1, B2, B3, NS_PLAIN, nullptr);
  float *Y = B3, *Z = B2, *T = B4, *F1 = B0, *F2 = B1;
  for (int i = 0; i < 3; ++i) {                 // iterN-2 = 3 loop iterations
    ns_gemm_kernel<<<gg, 128, 0, stream>>>(Z, Y, T, NS_F, nullptr);     // ZY
    ns_gemm_kernel<<<gg, 128, 0, stream>>>(Y, T, F1, NS_PLAIN, nullptr); // Y'
    ns_gemm_kernel<<<gg, 128, 0, stream>>>(T, Z, F2, NS_PLAIN, nullptr); // Z'
    float* oy = Y; float* oz = Z;
    Y = F1; Z = F2; F1 = oy; F2 = oz;
  }
  ns_gemm_kernel<<<gg, 128, 0, stream>>>(Z, Y, T, NS_F, nullptr);       // ZY
  ns_gemm_kernel<<<gg, 128, 0, stream>>>(Y, T, F1, NS_SCALE, sqrtn);    // cov_sqrt

  semlp_kernel<<<BATCH, 256, 0, stream>>>(F1, w1, b1, w2, b2, sgate);
  scale_kernel<<<(BATCH * CCH * MPIX / 4) / 256, 256, 0, stream>>>(x, sgate, out);
}
